// FullLocalTransBlock_89163521065542
// MI455X (gfx1250) — matrix operand order repaired and checked by reading, not compiled, not run on hardware
//
#include <hip/hip_runtime.h>

typedef _Float16 h16;
typedef __attribute__((ext_vector_type(2)))  _Float16 h2v;
typedef __attribute__((ext_vector_type(16))) _Float16 v16h;
typedef __attribute__((ext_vector_type(8)))  float    v8f;

// ---------------- problem constants ----------------
constexpr int Bc = 4, Cc = 64, Hc = 256, Wc = 256;
constexpr int HWfull = Hc * Wc;           // 65536
constexpr int Hd = 64, Wd = 64, HWd = Hd * Wd;   // 4096 (after /4 pool)
constexpr int Hs = 128, Ws = 128, Lss = Hs * Ws; // 16384 (after /2 pool)
constexpr int Nrow = Bc * Cc;             // 256 rows in attention 1
constexpr int KParts = 4;                 // split-K factor for scores GEMM
constexpr float EPSN = 1e-12f;

// ---------------- block reductions ----------------
__device__ inline float blk_sum(float v, float* red) {
  int t = threadIdx.x;
  red[t] = v; __syncthreads();
  for (int s = blockDim.x >> 1; s > 0; s >>= 1) {
    if (t < s) red[t] += red[t + s];
    __syncthreads();
  }
  float r = red[0]; __syncthreads();
  return r;
}
__device__ inline float blk_max(float v, float* red) {
  int t = threadIdx.x;
  red[t] = v; __syncthreads();
  for (int s = blockDim.x >> 1; s > 0; s >>= 1) {
    if (t < s) red[t] = fmaxf(red[t], red[t + s]);
    __syncthreads();
  }
  float r = red[0]; __syncthreads();
  return r;
}

// ---------------- WMMA fragment loaders (contiguous-k order k=16*(L/16)+i, same for A and B, wave32) -------
// A (16x32 f16): lane L holds row M=L%16; VGPR pair i holds K = (L/16)*16 +
// i*2 + {0,1}
__device__ inline v16h frag_a_rowmajor(const h16* __restrict__ src, int row,
                                       int ld, int kbase, int lane) {
  const int kgrp = (lane >> 4) << 3;
  const h16* p = src + row * ld + kbase;
  v16h a;
#pragma unroll
  for (int i = 0; i < 8; ++i) {
    int kp = (kgrp << 1) + (i << 1);
    h2v t = *(const h2v*)(p + kp);
    a[2 * i] = t.x;
    a[2 * i + 1] = t.y;
  }
  return a;
}
// B (32x16 f16): lane L holds column N=L%16; element i holds K=(L/16)*16+i.
// Case 1: B[K][N] = src[N_row * ld + K]   (multiply by src^T, K contiguous)
__device__ inline v16h frag_b_kcontig(const h16* __restrict__ src, int col,
                                      int ld, int kbase, int lane) {
  const int kg = (lane >> 4) << 4;
  const h16* p = src + col * ld + kbase + kg;
  v16h b;
#pragma unroll
  for (int i = 0; i < 8; ++i) {
    h2v t = *(const h2v*)(p + 2 * i);
    b[2 * i] = t.x;
    b[2 * i + 1] = t.y;
  }
  return b;
}
// Case 2: B[K][N] = src[(kbase+K) * ld + col]   (K strided by ld)
__device__ inline v16h frag_b_kstride(const h16* __restrict__ src, int col,
                                      int ld, int kbase, int lane) {
  const int kg = (lane >> 4) << 4;
  v16h b;
#pragma unroll
  for (int i = 0; i < 16; ++i) {
    b[i] = src[(kbase + kg + i) * ld + col];
  }
  return b;
}

// ================= kernel 1: 4x4 maxpool =================
__global__ void k_pool4(const float* __restrict__ x, float* __restrict__ xd) {
  int idx = blockIdx.x * 256 + threadIdx.x;          // over Nrow*HWd
  int bc = idx >> 12, l = idx & 4095;
  int hd = l >> 6, wd = l & 63;
  const float* p = x + bc * HWfull + (hd * 4) * Wc + wd * 4;
  float m = -3.4e38f;
#pragma unroll
  for (int r = 0; r < 4; ++r)
#pragma unroll
    for (int c = 0; c < 4; ++c) m = fmaxf(m, p[r * Wc + c]);
  xd[idx] = m;
}

// ================= kernel 2: noise row stats (Sum, SumSq per slot) =========
__global__ void k_noise_stats(const float* __restrict__ x,
                              const int* __restrict__ noise_idx,
                              float* __restrict__ nstat) {
  __shared__ float red[256];
  int slot = blockIdx.x;
  const float* row = x + (long)noise_idx[slot] * HWfull;
  float s = 0.f, s2 = 0.f;
  for (int t = threadIdx.x; t < HWfull; t += 256) {
    float v = row[t];
    s += v; s2 += v * v;
  }
  s = blk_sum(s, red);
  s2 = blk_sum(s2, red);
  if (threadIdx.x == 0) { nstat[slot] = s; nstat[8 + slot] = s2; }
}

// ================= kernel 3: qkv1 build (f16) =================
__global__ void k_qkv1(const float* __restrict__ xd,
                       const float* __restrict__ w1,
                       const float* __restrict__ b1,
                       h16* __restrict__ qkvh) {
  int idx = blockIdx.x * 256 + threadIdx.x;          // over 768*4096
  int r = idx >> 12, l = idx & 4095;
  int b = r / 192, j = r - b * 192;
  int ch = j / 3;
  float v = xd[(b * Cc + ch) * HWd + l] * w1[j] + b1[j];
  qkvh[idx] = (h16)v;
}

// ========== kernel 4: scores = q @ k^T (WMMA f16), split-K x4 ==========
__global__ void k_scores(const h16* __restrict__ qkvh,
                         float* __restrict__ scoresp) {
  int lane = threadIdx.x & 31;
  int part = blockIdx.x & 3;                          // split-K part
  int tile = (blockIdx.x >> 2) * 8 + (threadIdx.x >> 5);  // 256 tiles
  int i0 = (tile >> 4) << 4, n0 = (tile & 15) << 4;
  const h16* q = qkvh;                 // rows [0,256)
  const h16* k = qkvh + 256 * HWd;     // rows [256,512)
  int arow = i0 + (lane & 15);
  int brow = n0 + (lane & 15);
  int kstart = part * (HWd / KParts);
  int kend = kstart + (HWd / KParts);
  v8f acc = {};
  for (int kb = kstart; kb < kend; kb += 32) {
    v16h a = frag_a_rowmajor(q, arow, HWd, kb, lane);
    v16h b = frag_b_kcontig(k, brow, HWd, kb, lane);
    acc = __builtin_amdgcn_wmma_f32_16x16x32_f16(false, a, false, b,
                                                 (short)0, acc, false, false);
  }
  int rbase = i0 + ((lane >> 4) << 3);
  int colN = n0 + (lane & 15);
#pragma unroll
  for (int v = 0; v < 8; ++v)
    scoresp[(part * Nrow + rbase + v) * Nrow + colN] = acc[v];
}

// ====== kernel 5: sum split-K partials + masked softmax rows -> f16 ======
__global__ void k_softmax(const float* __restrict__ scoresp,
                          const int* __restrict__ labels,
                          h16* __restrict__ attnh) {
  __shared__ float red[256];
  int i = blockIdx.x, n = threadIdx.x;
  int li = labels[i];
  float s = 0.f;
#pragma unroll
  for (int p = 0; p < KParts; ++p) s += scoresp[(p * Nrow + i) * Nrow + n];
  bool ok = (li != -1) && (labels[n] == li);
  float sv = ok ? s : -1e30f;
  float mx = blk_max(sv, red);
  float e = expf(sv - mx);
  float Z = blk_sum(e, red);
  attnh[i * Nrow + n] = (h16)(e / Z);
}

// ================= kernel 6: att = attn @ v (WMMA f16), valid mask =========
__global__ void k_attv(const h16* __restrict__ attnh,
                       const h16* __restrict__ qkvh,
                       const int* __restrict__ labels,
                       float* __restrict__ att) {
  int lane = threadIdx.x & 31;
  int tile = blockIdx.x * 8 + (threadIdx.x >> 5);    // 16 x 256 tiles
  int i0 = (tile >> 8) << 4, l0 = (tile & 255) << 4;
  const h16* vm = qkvh + 512 * HWd;    // rows [512,768)
  int arow = i0 + (lane & 15);
  int col = l0 + (lane & 15);
  v8f acc = {};
  for (int kb = 0; kb < Nrow; kb += 32) {
    v16h a = frag_a_rowmajor(attnh, arow, Nrow, kb, lane);
    v16h b = frag_b_kstride(vm, col, HWd, kb, lane);
    acc = __builtin_amdgcn_wmma_f32_16x16x32_f16(false, a, false, b,
                                                 (short)0, acc, false, false);
  }
  int rbase = i0 + ((lane >> 4) << 3);
#pragma unroll
  for (int v = 0; v < 8; ++v) {
    int row = rbase + v;
    att[row * HWd + col] = (labels[row] == -1) ? 0.f : acc[v];
  }
}

// ================= kernel 7: sim[b,c] without materializing `up` ===========
// att row staged into LDS through the CDNA5 async global->LDS DMA path.
__global__ void k_simdot(const float* __restrict__ att,
                         const float* __restrict__ x,
                         const int* __restrict__ noise_idx,
                         const float* __restrict__ up_k,
                         const float* __restrict__ up_b,
                         const float* __restrict__ nstat,
                         float* __restrict__ simv) {
  __shared__ __align__(16) float srow[HWd];
  __shared__ float red[256];
  __shared__ float kp[16];
  int bc = blockIdx.x, b = bc >> 6, c = bc & 63;
  if (threadIdx.x < 16) kp[threadIdx.x] = up_k[c * 16 + threadIdx.x];
  // async copy: 16 KB, 128-bit per lane, ASYNCcnt-tracked (no VGPR transit)
  for (int t = threadIdx.x; t < (HWd / 4); t += 256) {
    unsigned lds_off = (unsigned)(uintptr_t)(&srow[t * 4]);
    unsigned long long ga =
        (unsigned long long)(uintptr_t)(att + (size_t)bc * HWd + t * 4);
    asm volatile("global_load_async_to_lds_b128 %0, %1, off"
                 :: "v"(lds_off), "v"(ga) : "memory");
  }
  asm volatile("s_wait_asynccnt 0" ::: "memory");
  __syncthreads();
  float so = 0.f, sq = 0.f;
  for (int t = threadIdx.x; t < HWd; t += 256) {
    float v = srow[t];
    so += v; sq += v * v;
  }
  const float* xr0 = x + (long)noise_idx[2 * b] * HWfull;
  const float* xr1 = x + (long)noise_idx[2 * b + 1] * HWfull;
  float ub = up_b[c];
  float d0 = 0.f, d1 = 0.f;
  for (int idx = threadIdx.x; idx < HWfull; idx += 256) {
    int hh = idx >> 8, ww = idx & 255;
    float upv = srow[(hh >> 2) * Wd + (ww >> 2)] * kp[(hh & 3) * 4 + (ww & 3)] + ub;
    d0 += upv * xr0[idx];
    d1 += upv * xr1[idx];
  }
  d0 = blk_sum(d0, red);
  d1 = blk_sum(d1, red);
  so = blk_sum(so, red);
  sq = blk_sum(sq, red);
  if (threadIdx.x == 0) {
    float sk = 0.f, sk2 = 0.f;
#pragma unroll
    for (int t = 0; t < 16; ++t) { sk += kp[t]; sk2 += kp[t] * kp[t]; }
    float nu2 = sk2 * sq + 2.f * ub * sk * so + (float)HWfull * ub * ub;
    float nu = sqrtf(fmaxf(nu2, 0.f));
    float nn0 = sqrtf(fmaxf(nstat[8 + 2 * b], 0.f));
    float nn1 = sqrtf(fmaxf(nstat[8 + 2 * b + 1], 0.f));
    float s = 0.5f * (d0 / (fmaxf(nu, EPSN) * fmaxf(nn0, EPSN)) +
                      d1 / (fmaxf(nu, EPSN) * fmaxf(nn1, EPSN)));
    simv[bc] = s;
  }
}

// ========= kernel 8: fused out_clu -> grouped 16x16 mix -> 2x2 pool ========
__global__ void k_gconv_pool(const float* __restrict__ x,
                             const float* __restrict__ att,
                             const float* __restrict__ up_k,
                             const float* __restrict__ up_b,
                             const float* __restrict__ simv,
                             const float* __restrict__ clu_w,
                             const float* __restrict__ clu_b,
                             float* __restrict__ xds) {
  __shared__ float oc[16][512];   // 16 in-channels x (2 rows * 256 cols)
  __shared__ float wsh[256], kpl[256];
  __shared__ float ubl[16], sml[16], cbl[16];
  int blk = blockIdx.x;
  int b = blk >> 9;              // / (4*128)
  int rem = blk & 511;
  int g = rem >> 7, hs = rem & 127;
  int tid = threadIdx.x;
  wsh[tid] = clu_w[g * 256 + tid];
  kpl[tid] = up_k[(g * 16 + (tid >> 4)) * 16 + (tid & 15)];
  if (tid < 16) {
    int c = g * 16 + tid;
    ubl[tid] = up_b[c];
    sml[tid] = 2.f - simv[b * Cc + c];
    cbl[tid] = clu_b[c];
  }
  __syncthreads();
  for (int t = tid; t < 8192; t += 256) {
    int i = t >> 9, rr = (t >> 8) & 1, ww = t & 255;
    int hh = 2 * hs + rr;
    int c = g * 16 + i;
    float ocv = sml[i] * x[((b * Cc + c) * Hc + hh) * Wc + ww] +
                att[(b * Cc + c) * HWd + (hh >> 2) * Wd + (ww >> 2)] *
                    kpl[i * 16 + (hh & 3) * 4 + (ww & 3)] +
                ubl[i];
    oc[i][rr * 256 + ww] = ocv;
  }
  __syncthreads();
  for (int t = tid; t < 2048; t += 256) {
    int o = t >> 7, wsc = t & 127;
    float m = -3.4e38f;
#pragma unroll
    for (int rr = 0; rr < 2; ++rr)
#pragma unroll
      for (int cc = 0; cc < 2; ++cc) {
        int col = 2 * wsc + cc;
        float acc = cbl[o];
#pragma unroll
        for (int i = 0; i < 16; ++i) acc += wsh[o * 16 + i] * oc[i][rr * 256 + col];
        m = fmaxf(m, acc);
      }
    xds[((b * Cc + g * 16 + o) * Hs + hs) * Ws + wsc] = m;
  }
}

// ===== kernel 9: per-head attention 2 -> oh[b,c], templated on head ========
// channel index j/3 repeats ~3x: only 6 distinct channels per q/k/v matrix.
template <int N>
__global__ void k_attn2_t(const float* __restrict__ xds,
                          const float* __restrict__ qkv_w2,
                          float* __restrict__ ohv) {
  __shared__ float red[256];
  __shared__ float accbuf[16 * 256];
  int b = blockIdx.x;
  constexpr int jb = N * 16;
  constexpr int loq = jb / 3, lok = (64 + jb) / 3, lov = (128 + jb) / 3;
  float wqk[16], wv[16];
#pragma unroll
  for (int cc = 0; cc < 16; ++cc) {
    wqk[cc] = qkv_w2[jb + cc] * qkv_w2[64 + jb + cc];
    wv[cc] = qkv_w2[128 + jb + cc];
  }
  const float* xb = xds + (long)b * Cc * Lss;
  // pass 1: max of sc
  float lmax = -3.4e38f;
  for (int l = threadIdx.x; l < Lss; l += 256) {
    float vq[6], vk[6];
#pragma unroll
    for (int j = 0; j < 6; ++j) {
      vq[j] = xb[(loq + j) * Lss + l];
      vk[j] = xb[(lok + j) * Lss + l];
    }
    float s = 0.f;
#pragma unroll
    for (int cc = 0; cc < 16; ++cc)
      s += wqk[cc] * vq[(jb + cc) / 3 - loq] * vk[(64 + jb + cc) / 3 - lok];
    lmax = fmaxf(lmax, s * 0.25f);
  }
  float mx = blk_max(lmax, red);
  // pass 2: recompute, exp-sum and weighted v accumulation
  float lz = 0.f;
  float acc[16];
#pragma unroll
  for (int cc = 0; cc < 16; ++cc) acc[cc] = 0.f;
  for (int l = threadIdx.x; l < Lss; l += 256) {
    float vq[6], vk[6], vv[6];
#pragma unroll
    for (int j = 0; j < 6; ++j) {
      vq[j] = xb[(loq + j) * Lss + l];
      vk[j] = xb[(lok + j) * Lss + l];
      vv[j] = xb[(lov + j) * Lss + l];
    }
    float s = 0.f;
#pragma unroll
    for (int cc = 0; cc < 16; ++cc)
      s += wqk[cc] * vq[(jb + cc) / 3 - loq] * vk[(64 + jb + cc) / 3 - lok];
    float e = expf(s * 0.25f - mx);
    lz += e;
#pragma unroll
    for (int cc = 0; cc < 16; ++cc)
      acc[cc] += e * vv[(128 + jb + cc) / 3 - lov] * wv[cc];
  }
  float Z = blk_sum(lz, red);
#pragma unroll
  for (int cc = 0; cc < 16; ++cc) accbuf[cc * 256 + threadIdx.x] = acc[cc];
  __syncthreads();
  if (threadIdx.x < 16) {
    float tot = 0.f;
    for (int t = 0; t < 256; ++t) tot += accbuf[threadIdx.x * 256 + t];
    ohv[b * Cc + N * 16 + threadIdx.x] = tot / Z;
  }
}

// ====== kernel 10: ortho scalar, ov = bais_w@oh + b, sim2 (rank-1 sim) =====
__global__ void k_head(const float* __restrict__ clu_w,
                       const float* __restrict__ bais_w,
                       const float* __restrict__ bais_b,
                       const float* __restrict__ ohv,
                       const float* __restrict__ nstat,
                       float* __restrict__ ovv,
                       float* __restrict__ sim2v,
                       float* __restrict__ orthov) {
  __shared__ float red[1024];
  int t = threadIdx.x;
  int g = t >> 8, o = (t >> 4) & 15, p = t & 15;
  float s = 0.f;
#pragma unroll
  for (int i = 0; i < 16; ++i)
    s += clu_w[g * 256 + o * 16 + i] * clu_w[g * 256 + p * 16 + i];
  float d = s - ((o == p) ? 1.f : 0.f);
  float msum = blk_sum(d * d, red);
  if (t == 0) orthov[0] = 1.f - msum * (1.f / 1024.f);
  if (t < 256) {
    int b = t >> 6, oo = t & 63;
    float acc = bais_b[oo];
#pragma unroll
    for (int c = 0; c < 64; ++c) acc += bais_w[oo * 64 + c] * ohv[b * Cc + c];
    ovv[t] = acc;
    float f = acc / fmaxf(fabsf(acc) * 256.0f, EPSN);   // sqrt(HW)=256
    float nn0 = sqrtf(fmaxf(nstat[8 + 2 * b], 0.f));
    float nn1 = sqrtf(fmaxf(nstat[8 + 2 * b + 1], 0.f));
    sim2v[t] = 0.5f * f * (nstat[2 * b] / fmaxf(nn0, EPSN) +
                           nstat[2 * b + 1] / fmaxf(nn1, EPSN));
  }
}

// ===== kernel 11: final: recompute out_clu, combine with rank-1 out2 =======
__global__ void k_final(const float* __restrict__ x,
                        const float* __restrict__ att,
                        const float* __restrict__ up_k,
                        const float* __restrict__ up_b,
                        const float* __restrict__ simv,
                        const float* __restrict__ ovv,
                        const float* __restrict__ sim2v,
                        const float* __restrict__ orthov,
                        float* __restrict__ out) {
  int idx = blockIdx.x * 256 + threadIdx.x;
  int bc = idx >> 16, pix = idx & 65535;
  int hh = pix >> 8, ww = pix & 255;
  int c = bc & 63;
  float oc = (2.f - simv[bc]) * x[idx] +
             att[bc * HWd + (hh >> 2) * Wd + (ww >> 2)] *
                 up_k[c * 16 + (hh & 3) * 4 + (ww & 3)] +
             up_b[c];
  out[idx] = (1.f + orthov[0]) * oc + ovv[bc] * (1.f - sim2v[bc]);
}

// =================== launcher ===================
extern "C" void kernel_launch(void* const* d_in, const int* in_sizes, int n_in,
                              void* d_out, int out_size, void* d_ws, size_t ws_size,
                              hipStream_t stream) {
  const float* x       = (const float*)d_in[0];
  const int*   labels  = (const int*)d_in[1];
  const int*   nidx    = (const int*)d_in[2];
  const float* qkv_w1  = (const float*)d_in[3];
  const float* qkv_b1  = (const float*)d_in[4];
  const float* up_k    = (const float*)d_in[5];
  const float* up_b    = (const float*)d_in[6];
  const float* clu_w   = (const float*)d_in[7];
  const float* clu_b   = (const float*)d_in[8];
  const float* qkv_w2  = (const float*)d_in[9];
  const float* bais_w  = (const float*)d_in[10];
  const float* bais_b  = (const float*)d_in[11];
  float* out = (float*)d_out;

  // workspace carve (~15.9 MB)
  char* base = (char*)d_ws;
  float* xd      = (float*)base;  base += (size_t)Nrow * HWd * 4;            // 4 MB
  h16*   qkvh    = (h16*)base;    base += (size_t)768 * HWd * 2;             // 6 MB
  float* scoresp = (float*)base;  base += (size_t)KParts * Nrow * Nrow * 4;  // 1 MB
  h16*   attnh   = (h16*)base;    base += (size_t)Nrow * Nrow * 2;           // 128 KB
  float* att     = (float*)base;  base += (size_t)Nrow * HWd * 4;            // 4 MB
  float* simv    = (float*)base;  base += 256 * 4;
  float* nstat   = (float*)base;  base += 16 * 4;
  float* ohv     = (float*)base;  base += 256 * 4;
  float* ovv     = (float*)base;  base += 256 * 4;
  float* sim2v   = (float*)base;  base += 256 * 4;
  float* orthov  = (float*)base;  base += 4;
  // x_ds (16 MB) staged in the tail of d_out; fully consumed by k_attn2
  // before k_final rewrites every element of d_out.
  float* xds = out + (size_t)12 * 1024 * 1024;      // elements [12M,16M) of 16.77M

  k_pool4      <<<4096, 256, 0, stream>>>(x, xd);
  k_noise_stats<<<   8, 256, 0, stream>>>(x, nidx, nstat);
  k_qkv1       <<<12288, 256, 0, stream>>>(xd, qkv_w1, qkv_b1, qkvh);
  k_scores     <<< 128, 256, 0, stream>>>(qkvh, scoresp);
  k_softmax    <<< 256, 256, 0, stream>>>(scoresp, labels, attnh);
  k_attv       <<< 512, 256, 0, stream>>>(attnh, qkvh, labels, att);
  k_simdot     <<< 256, 256, 0, stream>>>(att, x, nidx, up_k, up_b, nstat, simv);
  k_gconv_pool <<<2048, 256, 0, stream>>>(x, att, up_k, up_b, simv, clu_w, clu_b, xds);
  k_attn2_t<0> <<<   4, 256, 0, stream>>>(xds, qkv_w2, ohv);
  k_attn2_t<1> <<<   4, 256, 0, stream>>>(xds, qkv_w2, ohv);
  k_attn2_t<2> <<<   4, 256, 0, stream>>>(xds, qkv_w2, ohv);
  k_attn2_t<3> <<<   4, 256, 0, stream>>>(xds, qkv_w2, ohv);
  k_head       <<<   1, 1024, 0, stream>>>(clu_w, bais_w, bais_b, ohv, nstat, ovv, sim2v, orthov);
  k_final      <<<65536, 256, 0, stream>>>(x, att, up_k, up_b, simv, ovv, sim2v, orthov, out);
}
